// CGCNN_57973468561890
// MI455X (gfx1250) — compile-verified, matrix operand order checked by reading, not run on hardware
//
#include <hip/hip_runtime.h>
#include <hip/hip_bf16.h>
#include <math.h>

// ---------------------------------------------------------------------------
// CGCNN forward for MI455X (gfx1250), wave32 + WMMA bf16 (f32 accumulate).
//
// N=40000 nodes, E=400000 edges, G=256 graphs, FC=128, AIN=92, BINS=32.
// Dominant cost: two [E,384]@[384,128] GEMMs (78.6 GFLOP) -> v_wmma_f32_16x16x32_bf16.
// BatchNorm over E rows forces a 2-pass scheme; pre-BN activations are
// materialized in f32 in the workspace (~537 MB layout, documented below).
//
// CDNA5 data movement in k_edge_gemm:
//   - h[dst]/h[src] row gathers: GLOBAL_LOAD_ASYNC_TO_LDS_B128 (ASYNCcnt)
//   - contiguous ef sub-block:   TENSOR_LOAD_TO_LDS (TDM, TENSORcnt) with LDS
//     padding (64-DWORD interval, 128-DWORD pad) to match the 768B z-tile pitch
// Both are guarded by __has_builtin with the proven load/ds_store fallback.
// ---------------------------------------------------------------------------

#define NN   40000
#define NE   400000
#define NG   256
#define FCH  128
#define AIN  92
#define BINS 32
#define EPSV 1e-5f

typedef unsigned short u16;
typedef __attribute__((ext_vector_type(16))) __bf16        v16bf;
typedef __attribute__((ext_vector_type(8)))  float         v8f;
typedef __attribute__((ext_vector_type(4)))  unsigned int  u32x4;
typedef __attribute__((ext_vector_type(8)))  int           i32x8;
typedef __attribute__((ext_vector_type(4)))  int           i32x4;
typedef __attribute__((ext_vector_type(4)))  float         f32x4;

// Address-space-qualified vector types for the async-LDS builtin
// (clang prints AS1 as __device__, AS3 as __shared__).
typedef __attribute__((address_space(1))) i32x4 gAS_i32x4;
typedef __attribute__((address_space(3))) i32x4 lAS_i32x4;

#if __has_builtin(__builtin_amdgcn_global_load_async_to_lds_b128)
#define HAVE_ASYNC_LDS 1
#endif
#if __has_builtin(__builtin_amdgcn_tensor_load_to_lds) && \
    __has_builtin(__builtin_amdgcn_s_wait_tensorcnt)
#define HAVE_TDM 1
#endif

union Frag16 { v16bf v; u32x4 q[2]; };

__device__ __forceinline__ u16 f2bf(float f) {
  union { float f; unsigned u; } x; x.f = f;
  unsigned r = x.u + 0x7FFFu + ((x.u >> 16) & 1u);   // round-to-nearest-even
  return (u16)(r >> 16);
}

__device__ __forceinline__ v8f zero8() {
  v8f z = {0.f, 0.f, 0.f, 0.f, 0.f, 0.f, 0.f, 0.f};
  return z;
}

__device__ __forceinline__ v8f wmma_bf16(v16bf a, v16bf b, v8f c) {
  // D = A(16x32 bf16) x B(32x16 bf16) + C(16x16 f32)
  return __builtin_amdgcn_wmma_f32_16x16x32_bf16(false, a, false, b,
                                                 (short)0, c, false, false);
}

// A fragment (16 M x 32 K) from row-major u16 tile in LDS.
// ISA layout: lanes 0-15 -> M, elems 0..7 = K half*8..+7, elems 8..15 = K 16+half*8..+7
__device__ __forceinline__ v16bf load_afrag(const u16* base, int rowStride) {
  int lane = threadIdx.x & 31;
  int half = lane >> 4, m = lane & 15;
  const u16* p = base + m * rowStride + half * 8;
  Frag16 f;
  f.q[0] = *(const u32x4*)(p);
  f.q[1] = *(const u32x4*)(p + 16);
  return f.v;
}

// B fragment: pre-packed (k_pack) so each lane reads 32 contiguous bytes.
__device__ __forceinline__ v16bf load_bfrag(const u16* pw, int kt, int nt) {
  int lane = threadIdx.x & 31;
  const u32x4* p = (const u32x4*)(pw + ((((kt * 8 + nt) * 32) + lane) << 4));
  Frag16 f;
  f.q[0] = p[0];
  f.q[1] = p[1];
  return f.v;
}

#ifdef HAVE_ASYNC_LDS
// Direct global -> LDS async copy of 16 bf16 (32 B) per call site, per lane.
__device__ __forceinline__ void async_cp32(const u16* gsrc, u16* ldst) {
  __builtin_amdgcn_global_load_async_to_lds_b128(
      (gAS_i32x4*)gsrc, (lAS_i32x4*)ldst, 0, 0);
  __builtin_amdgcn_global_load_async_to_lds_b128(
      (gAS_i32x4*)(gsrc + 8), (lAS_i32x4*)(ldst + 8), 0, 0);
}
__device__ __forceinline__ void wait_async_all() {
#if __has_builtin(__builtin_amdgcn_s_wait_asynccnt)
  __builtin_amdgcn_s_wait_asynccnt(0);
#else
  asm volatile("s_wait_asynccnt 0x0" ::: "memory");
#endif
}
#endif

#ifdef HAVE_TDM
// TDM: load a 2D bf16 tile [64 rows x 128 elems] (row stride 128 elems in
// memory, i.e. fully contiguous here) into LDS with 512B padding after every
// 256B row -> lands on the 768B (384 u16) pitch of the z-tile's ef columns.
__device__ __forceinline__ void tdm_load_ef(const u16* efsrc, unsigned ldsByteAddr) {
  unsigned long long ga = (unsigned long long)(const void*)efsrc;
  u32x4 g0;
  g0[0] = 1u;                                   // count=1, user descriptor
  g0[1] = ldsByteAddr;                          // lds_addr (bytes)
  g0[2] = (unsigned)(ga & 0xFFFFFFFFu);         // global_addr[31:0]
  g0[3] = (unsigned)((ga >> 32) & 0x01FFFFFFu)  // global_addr[56:32]
          | (2u << 30);                         // type = 2 ("image")
  i32x8 g1;
  g1[0] = (int)((1u << 16)      // data_size = 2 bytes
              | (1u << 20)      // pad_enable
              | (5u << 22)      // pad_interval: 64 DWORDs (= 256B tile row)
              | (127u << 25));  // pad_amount: 128 DWORDs (= 512B pad)
  g1[1] = (int)(128u << 16);                        // tensor_dim0 lo16 = 128
  g1[2] = (int)(((unsigned)NE & 0xFFFFu) << 16);    // dim0 hi16=0 | dim1 lo16
  g1[3] = (int)(((unsigned)NE >> 16) | (128u << 16)); // dim1 hi16 | tile_dim0=128
  g1[4] = 64;                                       // tile_dim1=64, tile_dim2=0
  g1[5] = 128;                                      // tensor_dim0_stride lo32
  g1[6] = 0;
  g1[7] = 0;
  i32x4 zz = {0, 0, 0, 0};
#if defined(__clang_major__) && (__clang_major__ >= 23)
  i32x8 z8 = {0, 0, 0, 0, 0, 0, 0, 0};
  __builtin_amdgcn_tensor_load_to_lds(g0, g1, zz, zz, z8, 0);
#else
  __builtin_amdgcn_tensor_load_to_lds(g0, g1, zz, zz, 0);
#endif
}
#endif

// ---------------------------------------------------------------------------
// Pack a [Krows, 128] f32 weight into bf16 WMMA B-fragment order.
// idx = ((kt*8 + nt)*32 + lane)*16 + e ; rows beyond Krows are zero padding.
// ---------------------------------------------------------------------------
__global__ void __launch_bounds__(256) k_pack(const float* __restrict__ W,
                                              int Krows, int ktiles,
                                              u16* __restrict__ dst) {
  int idx = blockIdx.x * 256 + threadIdx.x;
  int total = ktiles * 4096;
  if (idx >= total) return;
  int e    = idx & 15;
  int lane = (idx >> 4) & 31;
  int nt   = (idx >> 9) & 7;
  int kt   = idx >> 12;
  int half = lane >> 4, n = lane & 15;
  int kk = (e < 8) ? (half * 8 + e) : (16 + half * 8 + (e - 8));
  int K = kt * 32 + kk;
  int col = nt * 16 + n;
  float v = (K < Krows) ? W[(size_t)K * FCH + col] : 0.f;
  dst[idx] = f2bf(v);
}

// ---------------------------------------------------------------------------
// h = x @ W_atom + b_atom  (K=92 padded to 96). Writes f32 and bf16 copies.
// ---------------------------------------------------------------------------
__global__ void __launch_bounds__(256) k_atom(const float* __restrict__ x,
                                              const float* __restrict__ b_atom,
                                              const u16* __restrict__ pwA,
                                              float* __restrict__ h,
                                              u16* __restrict__ hb) {
  __shared__ __align__(16) u16 xa[128][96];
  int tid = threadIdx.x;
  int row0 = blockIdx.x * 128;
  for (int idx = tid; idx < 128 * 96; idx += 256) {
    int r = idx / 96, k = idx - r * 96;
    int node = row0 + r;
    float v = (node < NN && k < AIN) ? x[(size_t)node * AIN + k] : 0.f;
    xa[r][k] = f2bf(v);
  }
  __syncthreads();
  int w = tid >> 5, lane = tid & 31;
  int mtile = blockIdx.x * 8 + w;
  if (mtile * 16 >= NN) return;
  int half = lane >> 4, n = lane & 15;
  for (int nt = 0; nt < 8; ++nt) {
    v8f acc = zero8();
#pragma unroll
    for (int kt = 0; kt < 3; ++kt) {
      v16bf a = load_afrag(&xa[w * 16][kt * 32], 96);
      v16bf b = load_bfrag(pwA, kt, nt);
      acc = wmma_bf16(a, b, acc);
    }
    int col = nt * 16 + n;
    float bias = b_atom[col];
#pragma unroll
    for (int r = 0; r < 8; ++r) {
      int node = mtile * 16 + half * 8 + r;
      float v = acc[r] + bias;
      h[(size_t)node * FCH + col]  = v;
      hb[(size_t)node * FCH + col] = f2bf(v);
    }
  }
}

// ---------------------------------------------------------------------------
// ef = RBF(edge_attr) @ W_edge + b_edge, stored bf16.  K=32 -> 1 WMMA/tile.
// centers = linspace(0,8,32); gamma = 1/(8/31) = 3.875
// ---------------------------------------------------------------------------
__global__ void __launch_bounds__(256) k_edgefeat(const float* __restrict__ ea,
                                                  const float* __restrict__ b_edge,
                                                  const u16* __restrict__ pwE,
                                                  u16* __restrict__ ef) {
  __shared__ __align__(16) u16 rb[128][32];
  __shared__ float dv[128];
  int tid = threadIdx.x;
  int e0 = blockIdx.x * 128;
  if (tid < 128) dv[tid] = ea[e0 + tid];
  __syncthreads();
  for (int idx = tid; idx < 128 * 32; idx += 256) {
    int r = idx >> 5, bin = idx & 31;
    float c = (float)bin * (8.0f / 31.0f);
    float d = dv[r] - c;
    rb[r][bin] = f2bf(__expf(-3.875f * d * d));
  }
  __syncthreads();
  int w = tid >> 5, lane = tid & 31;
  int half = lane >> 4, n = lane & 15;
  int etile = blockIdx.x * 8 + w;
  v16bf a = load_afrag(&rb[w * 16][0], 32);
  for (int nt = 0; nt < 8; ++nt) {
    v8f acc = zero8();
    v16bf b = load_bfrag(pwE, 0, nt);
    acc = wmma_bf16(a, b, acc);
    int col = nt * 16 + n;
    float bias = b_edge[col];
#pragma unroll
    for (int r = 0; r < 8; ++r) {
      int e = etile * 16 + half * 8 + r;
      ef[(size_t)e * FCH + col] = f2bf(acc[r] + bias);
    }
  }
}

// ---------------------------------------------------------------------------
// Main edge GEMMs: u_ei = z@W_ei, u_eu = z@W_eu with z=[h[dst]|h[src]|ef].
// (Linear biases b_ei/b_eu dropped: BN subtracts the mean -> they cancel.)
// Block = 64 edges x 128 cols; 8 waves = 4 M-tiles x 2 N-halves; K=384 (12 kt).
// Also accumulates per-column sum / sum-of-squares for BN (LDS -> global f32 atomics).
// ---------------------------------------------------------------------------
__global__ void __launch_bounds__(256) k_edge_gemm(const int* __restrict__ eidx,
                                                   const u16* __restrict__ hb,
                                                   const u16* __restrict__ ef,
                                                   const u16* __restrict__ pwEI,
                                                   const u16* __restrict__ pwEU,
                                                   float* __restrict__ uei,
                                                   float* __restrict__ ueu,
                                                   float* __restrict__ gstat) {
  __shared__ __align__(16) u16 z[64][384];   // 48 KB
  __shared__ float lstat[512];               // sum_ei | sq_ei | sum_eu | sq_eu
  int tid = threadIdx.x;
  for (int i = tid; i < 512; i += 256) lstat[i] = 0.f;
  int e0 = blockIdx.x * 64;

#ifdef HAVE_TDM
  // One TDM op per workgroup (wave 0 only; TDM ignores EXEC, issues per wave):
  // ef rows e0..e0+63 -> z[:, 256:384] via LDS padding.
  if (tid < 32) tdm_load_ef(ef + (size_t)e0 * FCH, (unsigned)(size_t)&z[0][256]);
  const int NSEG = 16;    // stage only the h[dst] | h[src] gathers below
#else
  const int NSEG = 24;    // stage everything
#endif

  for (int idx = tid; idx < 64 * NSEG; idx += 256) {
    int r = idx / NSEG, seg = idx - r * NSEG;   // 32B (16 bf16) per segment
    int e = e0 + r;
    const u16* sp;
    if (seg < 8) {                              // cols 0..127   : h[dst]
      int node = eidx[NE + e];
      sp = hb + (size_t)node * FCH + seg * 16;
    } else if (seg < 16) {                      // cols 128..255 : h[src]
      int node = eidx[e];
      sp = hb + (size_t)node * FCH + (seg - 8) * 16;
    } else {                                    // cols 256..383 : ef (fallback)
      sp = ef + (size_t)e * FCH + (seg - 16) * 16;
    }
    u16* dp = &z[r][seg * 16];
#ifdef HAVE_ASYNC_LDS
    async_cp32(sp, dp);
#else
    u32x4* d4 = (u32x4*)dp;
    const u32x4* s4 = (const u32x4*)sp;
    d4[0] = s4[0];
    d4[1] = s4[1];
#endif
  }
#ifdef HAVE_ASYNC_LDS
  wait_async_all();
#endif
#ifdef HAVE_TDM
  if (tid < 32) __builtin_amdgcn_s_wait_tensorcnt(0);
#endif
  __syncthreads();

  int w = tid >> 5, lane = tid & 31;
  int mt = w >> 1, nh = w & 1;
  int half = lane >> 4, n = lane & 15;
  v8f aei[4], aeu[4];
#pragma unroll
  for (int j = 0; j < 4; ++j) { aei[j] = zero8(); aeu[j] = zero8(); }

  for (int kt = 0; kt < 12; ++kt) {
    if (kt + 1 < 12) {   // global_prefetch_b8 on next K-tile's B fragments
      __builtin_prefetch(pwEI + (size_t)(((kt + 1) * 8 + nh * 4) * 32) * 16, 0, 3);
      __builtin_prefetch(pwEU + (size_t)(((kt + 1) * 8 + nh * 4) * 32) * 16, 0, 3);
    }
    v16bf a = load_afrag(&z[mt * 16][kt * 32], 384);
#pragma unroll
    for (int j = 0; j < 4; ++j) {
      int nt = nh * 4 + j;
      aei[j] = wmma_bf16(a, load_bfrag(pwEI, kt, nt), aei[j]);
      aeu[j] = wmma_bf16(a, load_bfrag(pwEU, kt, nt), aeu[j]);
    }
  }

#pragma unroll
  for (int j = 0; j < 4; ++j) {
    int col = (nh * 4 + j) * 16 + n;
    float s0 = 0.f, q0 = 0.f, s1 = 0.f, q1 = 0.f;
#pragma unroll
    for (int r = 0; r < 8; ++r) {
      int e = e0 + mt * 16 + half * 8 + r;
      float v0 = aei[j][r], v1 = aeu[j][r];
      uei[(size_t)e * FCH + col] = v0;
      ueu[(size_t)e * FCH + col] = v1;
      s0 += v0; q0 += v0 * v0; s1 += v1; q1 += v1 * v1;
    }
    unsafeAtomicAdd(&lstat[col],       s0);
    unsafeAtomicAdd(&lstat[128 + col], q0);
    unsafeAtomicAdd(&lstat[256 + col], s1);
    unsafeAtomicAdd(&lstat[384 + col], q1);
  }
  __syncthreads();
  for (int i = tid; i < 512; i += 256) unsafeAtomicAdd(&gstat[i], lstat[i]);
}

// ---------------------------------------------------------------------------
// BN parameter finalize: a = g*rsqrt(var+eps), b' = b - mu*a   (biased var)
// ---------------------------------------------------------------------------
__global__ void __launch_bounds__(128) k_bnparams(const float* __restrict__ sum,
                                                  const float* __restrict__ sq,
                                                  const float* __restrict__ g,
                                                  const float* __restrict__ b,
                                                  float invCnt,
                                                  float* __restrict__ oa,
                                                  float* __restrict__ ob) {
  int c = threadIdx.x;
  float mu  = sum[c] * invCnt;
  float var = sq[c] * invCnt - mu * mu;
  float a = g[c] * rsqrtf(var + EPSV);
  oa[c] = a;
  ob[c] = b[c] - mu * a;
}

// ---------------------------------------------------------------------------
// gate = sigmoid(bn(u_ei)), core = softplus(bn(u_eu)); scatter gate*core -> agg[dst].
// st layout: a_ei@0, b_ei@128, a_eu@256, b_eu@384
// ---------------------------------------------------------------------------
__global__ void __launch_bounds__(256) k_edge_post(const float* __restrict__ uei,
                                                   const float* __restrict__ ueu,
                                                   const float* __restrict__ st,
                                                   const int* __restrict__ eidx,
                                                   float* __restrict__ agg) {
  long gid = (long)blockIdx.x * 256 + threadIdx.x;   // E*32 threads, 4 cols each
  int e = (int)(gid >> 5);
  int col = ((int)gid & 31) * 4;
  int dst = eidx[NE + e];
  f32x4 u0 = *(const f32x4*)(uei + (size_t)e * FCH + col);
  f32x4 u1 = *(const f32x4*)(ueu + (size_t)e * FCH + col);
  f32x4 ag = *(const f32x4*)(st + col);
  f32x4 bg = *(const f32x4*)(st + 128 + col);
  f32x4 au = *(const f32x4*)(st + 256 + col);
  f32x4 bu = *(const f32x4*)(st + 384 + col);
#pragma unroll
  for (int i = 0; i < 4; ++i) {
    float xg = ag[i] * u0[i] + bg[i];
    float xc = au[i] * u1[i] + bu[i];
    float gate = 1.f / (1.f + __expf(-xg));
    float core = (xc > 20.f) ? xc : log1pf(__expf(xc));
    unsafeAtomicAdd(&agg[(size_t)dst * FCH + col + i], gate * core);
  }
}

// ---------------------------------------------------------------------------
// Column sum / sumsq of agg over N rows (for the agg BatchNorm).
// ---------------------------------------------------------------------------
__global__ void __launch_bounds__(256) k_agg_stats(const float* __restrict__ agg,
                                                   float* __restrict__ gsum,
                                                   float* __restrict__ gsq) {
  __shared__ float rs[256], rq[256];
  int t = threadIdx.x;
  int col = t & 127, half = t >> 7;
  int r0 = blockIdx.x * 32 + half * 16;
  float s = 0.f, q = 0.f;
#pragma unroll
  for (int r = 0; r < 16; ++r) {
    float v = agg[(size_t)(r0 + r) * FCH + col];
    s += v; q += v * v;
  }
  rs[t] = s; rq[t] = q;
  __syncthreads();
  if (t < 128) {
    unsafeAtomicAdd(&gsum[col], rs[t] + rs[t + 128]);
    unsafeAtomicAdd(&gsq[col],  rq[t] + rq[t + 128]);
  }
}

// ---------------------------------------------------------------------------
// h2 = (h + bn(agg)) @ W_atoms + b_atoms ; pooled: graphFeat[batch[n]] += h2[n]
// st: a_ag@0, b_ag@128
// ---------------------------------------------------------------------------
__global__ void __launch_bounds__(256) k_node_out(const float* __restrict__ h,
                                                  const float* __restrict__ agg,
                                                  const float* __restrict__ st,
                                                  const int* __restrict__ batch,
                                                  const float* __restrict__ b_atoms,
                                                  const u16* __restrict__ pwAT,
                                                  float* __restrict__ gf) {
  __shared__ __align__(16) u16 at[128][128];   // 32 KB
  __shared__ int bid[128];
  int tid = threadIdx.x;
  int row0 = blockIdx.x * 128;
  for (int idx = tid; idx < 128 * 128; idx += 256) {
    int r = idx >> 7, k = idx & 127;
    int node = row0 + r;
    float v = 0.f;
    if (node < NN)
      v = h[(size_t)node * FCH + k] +
          (st[k] * agg[(size_t)node * FCH + k] + st[128 + k]);
    at[r][k] = f2bf(v);
  }
  if (tid < 128) {
    int node = row0 + tid;
    bid[tid] = (node < NN) ? batch[node] : 0;
  }
  __syncthreads();
  int w = tid >> 5, lane = tid & 31;
  int mtile = blockIdx.x * 8 + w;
  if (mtile * 16 >= NN) return;
  int half = lane >> 4, n = lane & 15;
  for (int nt = 0; nt < 8; ++nt) {
    v8f acc = zero8();
#pragma unroll
    for (int kt = 0; kt < 4; ++kt) {
      v16bf a = load_afrag(&at[w * 16][kt * 32], 128);
      v16bf b = load_bfrag(pwAT, kt, nt);
      acc = wmma_bf16(a, b, acc);
    }
    int col = nt * 16 + n;
    float bias = b_atoms[col];
#pragma unroll
    for (int r = 0; r < 8; ++r) {
      int m = w * 16 + half * 8 + r;
      unsafeAtomicAdd(&gf[(size_t)bid[m] * FCH + col], acc[r] + bias);
    }
  }
}

__global__ void __launch_bounds__(256) k_counts(const int* __restrict__ batch,
                                                float* __restrict__ cnt) {
  int i = blockIdx.x * 256 + threadIdx.x;
  if (i < NN) unsafeAtomicAdd(&cnt[batch[i]], 1.0f);
}

// ---------------------------------------------------------------------------
// Per-graph readout: silu(mean_feat @ W_fc + b_fc) @ W_out + b_out
// ---------------------------------------------------------------------------
__global__ void __launch_bounds__(128) k_readout(const float* __restrict__ gf,
                                                 const float* __restrict__ cnt,
                                                 const float* __restrict__ Wfc,
                                                 const float* __restrict__ bfc,
                                                 const float* __restrict__ Wout,
                                                 const float* __restrict__ bout,
                                                 float* __restrict__ out) {
  __shared__ float feat[128];
  __shared__ float red[128];
  int g = blockIdx.x, t = threadIdx.x;
  feat[t] = gf[(size_t)g * FCH + t] / fmaxf(cnt[g], 1.0f);
  __syncthreads();
  float acc = bfc[t];
  for (int k = 0; k < 128; ++k) acc += feat[k] * Wfc[k * FCH + t];
  float sig = 1.f / (1.f + __expf(-acc));
  red[t] = (acc * sig) * Wout[t];
  __syncthreads();
  for (int s = 64; s > 0; s >>= 1) {
    if (t < s) red[t] += red[t + s];
    __syncthreads();
  }
  if (t == 0) out[g] = red[0] + bout[0];
}

// ---------------------------------------------------------------------------
extern "C" void kernel_launch(void* const* d_in, const int* in_sizes, int n_in,
                              void* d_out, int out_size, void* d_ws, size_t ws_size,
                              hipStream_t stream) {
  const float* x       = (const float*)d_in[0];
  const float* ea      = (const float*)d_in[1];
  const int*   eidx    = (const int*)d_in[2];
  const int*   batch   = (const int*)d_in[3];
  const float* W_atom  = (const float*)d_in[4];
  const float* b_atom  = (const float*)d_in[5];
  const float* W_edge  = (const float*)d_in[6];
  const float* b_edge  = (const float*)d_in[7];
  const float* W_ei    = (const float*)d_in[8];
  const float* gei     = (const float*)d_in[10];
  const float* bei     = (const float*)d_in[11];
  const float* W_eu    = (const float*)d_in[12];
  const float* geu     = (const float*)d_in[14];
  const float* beu     = (const float*)d_in[15];
  const float* bng     = (const float*)d_in[16];
  const float* bnb     = (const float*)d_in[17];
  const float* W_atoms = (const float*)d_in[18];
  const float* b_atoms = (const float*)d_in[19];
  const float* W_fc    = (const float*)d_in[20];
  const float* b_fc    = (const float*)d_in[21];
  const float* W_out   = (const float*)d_in[22];
  const float* b_out   = (const float*)d_in[23];
  float* out = (float*)d_out;
  char* ws = (char*)d_ws;

  // Workspace layout (all offsets 256B-aligned), total ~537.5 MiB
  const size_t OFF_H    = 0;                                  // h f32      20.48 MB
  const size_t OFF_HB   = OFF_H    + (size_t)NN * FCH * 4;    // h bf16     10.24 MB
  const size_t OFF_EF   = OFF_HB   + (size_t)NN * FCH * 2;    // ef bf16   102.40 MB
  const size_t OFF_UEI  = OFF_EF   + (size_t)NE * FCH * 2;    // u_ei f32  204.80 MB
  const size_t OFF_UEU  = OFF_UEI  + (size_t)NE * FCH * 4;    // u_eu f32  204.80 MB
  const size_t OFF_AGG  = OFF_UEU  + (size_t)NE * FCH * 4;    // agg f32    20.48 MB
  const size_t OFF_GF   = OFF_AGG  + (size_t)NN * FCH * 4;    // graphFeat 131 KB
  const size_t OFF_CNT  = OFF_GF   + (size_t)NG * FCH * 4;    // counts      1 KB
  const size_t OFF_ST   = OFF_CNT  + 1024;                    // stats       6 KB
  const size_t OFF_PWA  = OFF_ST   + 12 * 128 * 4;            // packed W_atom  (3 kt)
  const size_t OFF_PWE  = OFF_PWA  + 3  * 4096 * 2;           // packed W_edge  (1 kt)
  const size_t OFF_PWEI = OFF_PWE  + 1  * 4096 * 2;           // packed W_ei   (12 kt)
  const size_t OFF_PWEU = OFF_PWEI + 12 * 4096 * 2;           // packed W_eu   (12 kt)
  const size_t OFF_PWAT = OFF_PWEU + 12 * 4096 * 2;           // packed W_atoms (4 kt)

  float* hF   = (float*)(ws + OFF_H);
  u16*   hB   = (u16*)  (ws + OFF_HB);
  u16*   efB  = (u16*)  (ws + OFF_EF);
  float* uei  = (float*)(ws + OFF_UEI);
  float* ueu  = (float*)(ws + OFF_UEU);
  float* aggF = (float*)(ws + OFF_AGG);
  float* gfF  = (float*)(ws + OFF_GF);
  float* cntF = (float*)(ws + OFF_CNT);
  float* st   = (float*)(ws + OFF_ST);
  // st: 0 sum_ei | 128 sq_ei | 256 sum_eu | 384 sq_eu | 512 sum_ag | 640 sq_ag
  //     768 a_ei | 896 b_ei  | 1024 a_eu  | 1152 b_eu | 1280 a_ag  | 1408 b_ag
  u16* pwA  = (u16*)(ws + OFF_PWA);
  u16* pwE  = (u16*)(ws + OFF_PWE);
  u16* pwEI = (u16*)(ws + OFF_PWEI);
  u16* pwEU = (u16*)(ws + OFF_PWEU);
  u16* pwAT = (u16*)(ws + OFF_PWAT);

  // Zero atomic-accumulated buffers (ws is poisoned once by the harness).
  (void)hipMemsetAsync(ws + OFF_AGG, 0, (size_t)NN * FCH * 4, stream);
  (void)hipMemsetAsync(ws + OFF_GF,  0, (size_t)NG * FCH * 4, stream);
  (void)hipMemsetAsync(ws + OFF_CNT, 0, NG * 4, stream);
  (void)hipMemsetAsync(ws + OFF_ST,  0, 6 * 128 * 4, stream);

  // Weight packing (bf16 fragment order)
  k_pack<<<(3 * 4096) / 256, 256, 0, stream>>>(W_atom, AIN, 3, pwA);
  k_pack<<<(1 * 4096) / 256, 256, 0, stream>>>(W_edge, BINS, 1, pwE);
  k_pack<<<(12 * 4096) / 256, 256, 0, stream>>>(W_ei, 3 * FCH, 12, pwEI);
  k_pack<<<(12 * 4096) / 256, 256, 0, stream>>>(W_eu, 3 * FCH, 12, pwEU);
  k_pack<<<(4 * 4096) / 256, 256, 0, stream>>>(W_atoms, FCH, 4, pwAT);

  // Pipeline
  k_atom<<<313, 256, 0, stream>>>(x, b_atom, pwA, hF, hB);
  k_edgefeat<<<NE / 128, 256, 0, stream>>>(ea, b_edge, pwE, efB);
  k_edge_gemm<<<NE / 64, 256, 0, stream>>>(eidx, hB, efB, pwEI, pwEU, uei, ueu, st);
  k_bnparams<<<1, 128, 0, stream>>>(st,       st + 128, gei, bei, 1.0f / NE, st + 768,  st + 896);
  k_bnparams<<<1, 128, 0, stream>>>(st + 256, st + 384, geu, beu, 1.0f / NE, st + 1024, st + 1152);
  k_edge_post<<<(NE * 32) / 256, 256, 0, stream>>>(uei, ueu, st + 768, eidx, aggF);
  k_agg_stats<<<NN / 32, 256, 0, stream>>>(aggF, st + 512, st + 640);
  k_bnparams<<<1, 128, 0, stream>>>(st + 512, st + 640, bng, bnb, 1.0f / NN, st + 1280, st + 1408);
  k_node_out<<<313, 256, 0, stream>>>(hF, aggF, st + 1280, batch, b_atoms, pwAT, gfF);
  k_counts<<<(NN + 255) / 256, 256, 0, stream>>>(batch, cntF);
  k_readout<<<NG, 128, 0, stream>>>(gfF, cntF, W_fc, b_fc, W_out, b_out, out);
}
